// Attention_23493471109327
// MI455X (gfx1250) — matrix operand order repaired and checked by reading, not compiled, not run on hardware
//
#include <hip/hip_runtime.h>
#include <hip/hip_bf16.h>

#define BB 8
#define FF 512
#define TT 2048
#define AA 512

typedef __attribute__((ext_vector_type(16))) __bf16 v16bf;
typedef __attribute__((ext_vector_type(8)))  __bf16 v8bf;
typedef __attribute__((ext_vector_type(8)))  float  v8f;

// ---------------- fragment helpers (CDNA5 wave32 WMMA layouts) ----------------
// A-matrix 16x32 bf16: lane l -> row m = l&15; k = 16*(l>>4) .. +15,
// 16 contiguous bf16 loaded as two 8-element (16B) chunks.
__device__ __forceinline__ v16bf load_a_frag(const __bf16* p, int ld, int lane) {
    const __bf16* q = p + (size_t)(lane & 15) * ld + ((lane >> 4) << 4);
    v8bf lo = *(const v8bf*)q;
    v8bf hi = *(const v8bf*)(q + 8);
    v16bf r;
#pragma unroll
    for (int i = 0; i < 8; ++i) { r[i] = lo[i]; r[i + 8] = hi[i]; }
    return r;
}
// B-matrix 32x16 bf16 from an [N,K] row-major source: lane l -> column n = l&15,
// k = 16*(l>>4) .. +15, 16 contiguous bf16 (32B).
__device__ __forceinline__ v16bf load_b_frag(const __bf16* p, int ld, int lane) {
    const __bf16* q = p + (size_t)(lane & 15) * ld + ((lane >> 4) << 4);
    return *(const v16bf*)q;
}
__device__ __forceinline__ v8f wmma_bf16(v16bf a, v16bf b, v8f c) {
    return __builtin_amdgcn_wmma_f32_16x16x32_bf16(false, a, false, b, (short)0, c,
                                                   false, false);
}

// ---------------- weight packing ----------------
__global__ void k_pack_w(const float* __restrict__ W0, const float* __restrict__ W1,
                         const float* __restrict__ W2, const float* __restrict__ W3,
                         const float* __restrict__ W4, const float* __restrict__ W5,
                         __bf16* __restrict__ Wqkv) {
    int id = blockIdx.x * blockDim.x + threadIdx.x;
    if (id >= 3072 * 512) return;
    int n = id >> 9, k = id & 511;
    int h = n / 1536, r2 = n % 1536, type = r2 >> 9, a = r2 & 511;
    const float* src;
    switch (h * 3 + type) {
        case 0: src = W0; break;
        case 1: src = W1; break;
        case 2: src = W2; break;
        case 3: src = W3; break;
        case 4: src = W4; break;
        default: src = W5; break;
    }
    Wqkv[id] = (__bf16)src[(size_t)a * 512 + k];
}

__global__ void k_pack_b(const float* __restrict__ b0, const float* __restrict__ b1,
                         const float* __restrict__ b2, const float* __restrict__ b3,
                         const float* __restrict__ b4, const float* __restrict__ b5,
                         float* __restrict__ bqkv) {
    int n = blockIdx.x * blockDim.x + threadIdx.x;
    if (n >= 3072) return;
    int h = n / 1536, r2 = n % 1536, type = r2 >> 9, a = r2 & 511;
    const float* src;
    switch (h * 3 + type) {
        case 0: src = b0; break;
        case 1: src = b1; break;
        case 2: src = b2; break;
        case 3: src = b3; break;
        case 4: src = b4; break;
        default: src = b5; break;
    }
    bqkv[n] = src[a];
}

__global__ void k_pack_wp(const float* __restrict__ Wp, __bf16* __restrict__ WpB) {
    int id = blockIdx.x * blockDim.x + threadIdx.x;
    if (id < 512 * 1024) WpB[id] = (__bf16)Wp[id];
}

// ---------------- x[B,F,T] f32 -> xt[B*T, F] bf16 ----------------
__global__ void k_transpose_cast(const float* __restrict__ x, __bf16* __restrict__ xt) {
    __shared__ float tile[32][33];
    int b = blockIdx.z;
    int t0 = blockIdx.x * 32, f0 = blockIdx.y * 32;
    int tx = threadIdx.x, ty = threadIdx.y;  // 32 x 8
    const float* src = x + (size_t)b * FF * TT;
#pragma unroll
    for (int i = 0; i < 4; ++i) {
        int f = f0 + ty + i * 8;
        tile[ty + i * 8][tx] = src[(size_t)f * TT + t0 + tx];
    }
    __syncthreads();
    __bf16* dst = xt + (size_t)b * TT * FF;
#pragma unroll
    for (int i = 0; i < 4; ++i) {
        int t = t0 + ty + i * 8;
        dst[(size_t)t * FF + f0 + tx] = (__bf16)tile[tx][ty + i * 8];
    }
}

// ---------------- QKV GEMM: [16384,512] x [3072,512]^T, epilogue splits Q/K/V^T ---
// block = 128 threads (4 waves); wave tile = 32 rows x 64 cols (acc[2][4]).
__global__ void k_gemm_qkv(const __bf16* __restrict__ xt, const __bf16* __restrict__ Wqkv,
                           const float* __restrict__ bqkv, __bf16* __restrict__ Qb,
                           __bf16* __restrict__ Kb, __bf16* __restrict__ VT) {
    int lane = threadIdx.x & 31, wave = threadIdx.x >> 5;
    int m0 = blockIdx.y * 32;
    int n0 = blockIdx.x * 256 + wave * 64;
    v8f acc[2][4] = {};
#pragma unroll
    for (int kc = 0; kc < 16; ++kc) {
        v16bf a0 = load_a_frag(xt + (size_t)m0 * 512 + kc * 32, 512, lane);
        v16bf a1 = load_a_frag(xt + (size_t)(m0 + 16) * 512 + kc * 32, 512, lane);
#pragma unroll
        for (int j = 0; j < 4; ++j) {
            v16bf bfr = load_b_frag(Wqkv + (size_t)(n0 + j * 16) * 512 + kc * 32, 512, lane);
            acc[0][j] = wmma_bf16(a0, bfr, acc[0][j]);
            acc[1][j] = wmma_bf16(a1, bfr, acc[1][j]);
        }
    }
    int half = lane >> 4, col = lane & 15;
    int b = m0 >> 11;
#pragma unroll
    for (int i = 0; i < 2; ++i) {
        int mbase = m0 + i * 16;
#pragma unroll
        for (int j = 0; j < 4; ++j) {
            int n = n0 + j * 16 + col;
            float bias = bqkv[n];
            int h = n / 1536, r2 = n % 1536, type = r2 >> 9, a = r2 & 511;
            int bh = b * 2 + h;
#pragma unroll
            for (int r = 0; r < 8; ++r) {
                int m = mbase + r + 8 * half;
                int t = m & (TT - 1);
                float v = acc[i][j][r] + bias;
                if (type == 0)      Qb[((size_t)bh * TT + t) * AA + a] = (__bf16)v;
                else if (type == 1) Kb[((size_t)bh * TT + t) * AA + a] = (__bf16)v;
                else                VT[((size_t)bh * AA + a) * TT + t] = (__bf16)v;
            }
        }
    }
}

// ---------------- fused flash attention (cooperative, no redundant S) ----------
// block = 128 threads (4 waves) handling (bh, 16 query rows); s-step = 64.
// Each wave computes a distinct 16-wide S column tile; softmax stats and the
// bf16 P tile are merged through LDS; each wave owns a 128-wide A-slice of O.
__global__ void k_attention(const __bf16* __restrict__ Qb, const __bf16* __restrict__ Kb,
                            const __bf16* __restrict__ VT, __bf16* __restrict__ mh) {
    __shared__ __align__(16) __bf16 pbuf[16 * 64];
    __shared__ float smax[4][16];
    __shared__ float ssum[4][16];
    int lane = threadIdx.x & 31, wave = threadIdx.x >> 5;
    const int tiles = TT / 16;
    int bh = blockIdx.x / tiles;
    int m0 = (blockIdx.x % tiles) * 16;
    int h = bh & 1;
    const __bf16* Q = Qb + (size_t)bh * TT * AA;
    const __bf16* K = Kb + (size_t)bh * TT * AA;
    const __bf16* V = VT + (size_t)bh * AA * TT;
    int half = lane >> 4, col = lane & 15;

    v16bf qf[16];
#pragma unroll
    for (int kc = 0; kc < 16; ++kc)
        qf[kc] = load_a_frag(Q + (size_t)m0 * AA + kc * 32, AA, lane);

    v8f o[8] = {};
    float mrun[8], lrun[8];
#pragma unroll
    for (int r = 0; r < 8; ++r) { mrun[r] = -1e30f; lrun[r] = 0.0f; }

    const float kLog2e = 1.44269504088896340736f;
    const float scale  = 0.04419417382415922f;  // 1/sqrt(512)

    for (int st = 0; st < TT / 64; ++st) {
        int s0 = st * 64;
        int sb = s0 + wave * 16;  // this wave's S column tile
        if (s0 + 64 < TT) {       // prefetch next s-tile (global_prefetch_b8)
            __builtin_prefetch(K + (size_t)(s0 + 64 + wave * 16 + col) * AA + half * 32, 0, 3);
            __builtin_prefetch(V + (size_t)(wave * 128 + col) * TT + s0 + 64, 0, 3);
        }
        // --- S tile [16 rows x 16 cols] for this wave ---
        v8f sa = {};
#pragma unroll
        for (int kc = 0; kc < 16; ++kc) {
            v16bf kb = load_b_frag(K + (size_t)sb * AA + kc * 32, AA, lane);
            sa = wmma_bf16(qf[kc], kb, sa);
        }
        // --- per-tile row max -> LDS ---
        float p[8];
#pragma unroll
        for (int r = 0; r < 8; ++r) {
            p[r] = sa[r] * scale;
            float mx = p[r];
            mx = fmaxf(mx, __shfl_xor(mx, 1, 32));
            mx = fmaxf(mx, __shfl_xor(mx, 2, 32));
            mx = fmaxf(mx, __shfl_xor(mx, 4, 32));
            mx = fmaxf(mx, __shfl_xor(mx, 8, 32));
            if (col == 0) smax[wave][r + 8 * half] = mx;
        }
        __syncthreads();
        // --- merge maxes, exponentiate, publish P + partial sums ---
        float corr[8];
#pragma unroll
        for (int r = 0; r < 8; ++r) {
            int row = r + 8 * half;
            float m = mrun[r];
            m = fmaxf(m, smax[0][row]);
            m = fmaxf(m, smax[1][row]);
            m = fmaxf(m, smax[2][row]);
            m = fmaxf(m, smax[3][row]);
            corr[r] = exp2f((mrun[r] - m) * kLog2e);
            mrun[r] = m;
            p[r] = exp2f((p[r] - m) * kLog2e);
            float s = p[r];
            s += __shfl_xor(s, 1, 32);
            s += __shfl_xor(s, 2, 32);
            s += __shfl_xor(s, 4, 32);
            s += __shfl_xor(s, 8, 32);
            if (col == 0) ssum[wave][row] = s;
            pbuf[row * 64 + wave * 16 + col] = (__bf16)p[r];
        }
        __syncthreads();
        // --- update running sum, rescale O, accumulate P x V^T ---
#pragma unroll
        for (int r = 0; r < 8; ++r) {
            int row = r + 8 * half;
            float tot = ssum[0][row] + ssum[1][row] + ssum[2][row] + ssum[3][row];
            lrun[r] = lrun[r] * corr[r] + tot;
        }
#pragma unroll
        for (int nt = 0; nt < 8; ++nt)
#pragma unroll
            for (int r = 0; r < 8; ++r) o[nt][r] *= corr[r];
        v16bf pf0 = load_a_frag(pbuf, 64, lane);        // k = s0 .. s0+31
        v16bf pf1 = load_a_frag(pbuf + 32, 64, lane);   // k = s0+32 .. s0+63
#pragma unroll
        for (int nt = 0; nt < 8; ++nt) {
            const __bf16* vrow = V + (size_t)(wave * 128 + nt * 16) * TT + s0;
            o[nt] = wmma_bf16(pf0, load_b_frag(vrow, TT, lane), o[nt]);
            o[nt] = wmma_bf16(pf1, load_b_frag(vrow + 32, TT, lane), o[nt]);
        }
        __syncthreads();  // protect pbuf/smax/ssum for next iteration
    }
    float inv[8];
#pragma unroll
    for (int r = 0; r < 8; ++r) inv[r] = 1.0f / lrun[r];
    int bidx = bh >> 1;
#pragma unroll
    for (int nt = 0; nt < 8; ++nt) {
#pragma unroll
        for (int r = 0; r < 8; ++r) {
            int row  = m0 + r + 8 * half;
            int colg = h * 512 + wave * 128 + nt * 16 + col;
            mh[((size_t)bidx * TT + row) * 1024 + colg] = (__bf16)(o[nt][r] * inv[r]);
        }
    }
}

// ---------------- output GEMM: [16384,1024] x [512,1024]^T + bp -> y[B,F,T] f32 ----
__global__ void k_gemm_out(const __bf16* __restrict__ mh, const __bf16* __restrict__ WpB,
                           const float* __restrict__ bp, float* __restrict__ out) {
    int lane = threadIdx.x & 31, wave = threadIdx.x >> 5;
    int m0 = blockIdx.y * 32;
    int n0 = blockIdx.x * 256 + wave * 64;
    v8f acc[2][4] = {};
#pragma unroll
    for (int kc = 0; kc < 32; ++kc) {
        v16bf a0 = load_a_frag(mh + (size_t)m0 * 1024 + kc * 32, 1024, lane);
        v16bf a1 = load_a_frag(mh + (size_t)(m0 + 16) * 1024 + kc * 32, 1024, lane);
#pragma unroll
        for (int j = 0; j < 4; ++j) {
            v16bf bfr = load_b_frag(WpB + (size_t)(n0 + j * 16) * 1024 + kc * 32, 1024, lane);
            acc[0][j] = wmma_bf16(a0, bfr, acc[0][j]);
            acc[1][j] = wmma_bf16(a1, bfr, acc[1][j]);
        }
    }
    int half = lane >> 4, col = lane & 15;
    int b = m0 >> 11;
#pragma unroll
    for (int i = 0; i < 2; ++i) {
        int tbase = (m0 & (TT - 1)) + i * 16;
#pragma unroll
        for (int j = 0; j < 4; ++j) {
            int n = n0 + j * 16 + col;
            float bias = bp[n];
#pragma unroll
            for (int r = 0; r < 8; ++r) {
                int t = tbase + r + 8 * half;
                out[((size_t)b * FF + n) * TT + t] = acc[i][j][r] + bias;
            }
        }
    }
}

// ---------------- launcher ----------------
extern "C" void kernel_launch(void* const* d_in, const int* in_sizes, int n_in,
                              void* d_out, int out_size, void* d_ws, size_t ws_size,
                              hipStream_t stream) {
    (void)in_sizes; (void)n_in; (void)out_size; (void)ws_size;
    const float* x   = (const float*)d_in[0];
    const float* Wq1 = (const float*)d_in[1];  const float* bq1 = (const float*)d_in[2];
    const float* Wk1 = (const float*)d_in[3];  const float* bk1 = (const float*)d_in[4];
    const float* Wv1 = (const float*)d_in[5];  const float* bv1 = (const float*)d_in[6];
    const float* Wq2 = (const float*)d_in[7];  const float* bq2 = (const float*)d_in[8];
    const float* Wk2 = (const float*)d_in[9];  const float* bk2 = (const float*)d_in[10];
    const float* Wv2 = (const float*)d_in[11]; const float* bv2 = (const float*)d_in[12];
    const float* Wp  = (const float*)d_in[13]; const float* bp  = (const float*)d_in[14];
    float* out = (float*)d_out;

    const size_t M = (size_t)BB * TT;  // 16384
    char* ws = (char*)d_ws;
    size_t off = 0;
    auto carve = [&](size_t bytes) -> void* {
        off = (off + 255) & ~(size_t)255;
        void* p = ws + off;
        off += bytes;
        return p;
    };
    __bf16* xt   = (__bf16*)carve(M * FF * sizeof(__bf16));
    __bf16* Wqkv = (__bf16*)carve((size_t)3072 * 512 * sizeof(__bf16));
    float*  bqkv = (float*) carve(3072 * sizeof(float));
    __bf16* WpB  = (__bf16*)carve((size_t)512 * 1024 * sizeof(__bf16));
    __bf16* Qb   = (__bf16*)carve((size_t)BB * 2 * TT * AA * sizeof(__bf16));
    __bf16* Kb   = (__bf16*)carve((size_t)BB * 2 * TT * AA * sizeof(__bf16));
    __bf16* VT   = (__bf16*)carve((size_t)BB * 2 * AA * TT * sizeof(__bf16));
    __bf16* mh   = (__bf16*)carve(M * 1024 * sizeof(__bf16));

    k_pack_w<<<(3072 * 512 + 255) / 256, 256, 0, stream>>>(Wq1, Wk1, Wv1, Wq2, Wk2, Wv2, Wqkv);
    k_pack_b<<<(3072 + 255) / 256, 256, 0, stream>>>(bq1, bk1, bv1, bq2, bk2, bv2, bqkv);
    k_pack_wp<<<(512 * 1024 + 255) / 256, 256, 0, stream>>>(Wp, WpB);
    k_transpose_cast<<<dim3(TT / 32, FF / 32, BB), dim3(32, 8), 0, stream>>>(x, xt);
    k_gemm_qkv<<<dim3(3072 / 256, M / 32), 128, 0, stream>>>(xt, Wqkv, bqkv, Qb, Kb, VT);
    k_attention<<<BB * 2 * (TT / 16), 128, 0, stream>>>(Qb, Kb, VT, mh);
    k_gemm_out<<<dim3(512 / 256, M / 32), 128, 0, stream>>>(mh, WpB, bp, out);
}
